// MultiheadKANAttention_21500606283844
// MI455X (gfx1250) — matrix operand order repaired and checked by reading, not compiled, not run on hardware
//
#include <hip/hip_runtime.h>
#include <hip/hip_bf16.h>

typedef __bf16 bf16;
typedef __attribute__((ext_vector_type(16))) __bf16 v16bf;
typedef __attribute__((ext_vector_type(8)))  __bf16 v8bf;
typedef __attribute__((ext_vector_type(8)))  float  v8f;

#define HIDDEN 1024
#define NTOK   4096      // B*S
#define NOUT   3072
#define SEQ    2048
#define NHEAD  16
#define HDIM   64
#define KACT   9216      // 1024 silu + 8192 spline bases

__device__ __forceinline__ v8f wmma_bf16(v16bf a, v16bf b, v8f c) {
  return __builtin_amdgcn_wmma_f32_16x16x32_bf16(false, a, false, b, (short)0, c, false, false);
}

__device__ __forceinline__ float silu(float v) {
  return v / (1.0f + __expf(-v));
}

__device__ __forceinline__ v8bf cvt8(v8f v) {
  v8bf r;
#pragma unroll
  for (int j = 0; j < 8; ++j) r[j] = (bf16)v[j];
  return r;
}
__device__ __forceinline__ v16bf cat16(v8bf lo, v8bf hi) {
  return __builtin_shufflevector(lo, hi, 0,1,2,3,4,5,6,7,8,9,10,11,12,13,14,15);
}

// Closed-form cubic B-splines on the uniform grid (knots t_m = 0.4m - 2.2).
// Only bases j = i-3..i are nonzero where i = interval index; weights are the
// standard uniform cubic polynomials. Exact match of the Cox-de Boor reference
// (all 8 output bases are boundary-complete on this knot vector).
__device__ __forceinline__ void bspline8(float x, float* o) {
  float u = (x + 2.2f) * 2.5f;
  float fi = floorf(u);
  int i = (int)fi;
  float f = u - fi;
  float f2 = f * f, f3 = f2 * f;
  float om = 1.0f - f;
  float w3 = f3 * (1.0f / 6.0f);                                // j = i
  float w2 = (-3.0f * f3 + 3.0f * f2 + 3.0f * f + 1.0f) * (1.0f / 6.0f); // i-1
  float w1 = (3.0f * f3 - 6.0f * f2 + 4.0f) * (1.0f / 6.0f);    // j = i-2
  float w0 = om * om * om * (1.0f / 6.0f);                      // j = i-3
#pragma unroll
  for (int j = 0; j < 8; ++j) {
    int d = i - j;
    float v = 0.0f;
    v = (d == 0) ? w3 : v;
    v = (d == 1) ? w2 : v;
    v = (d == 2) ? w1 : v;
    v = (d == 3) ? w0 : v;
    o[j] = v;
  }
}

// ---------------------------------------------------------------------------
// Kernel 0: activation precompute. actF[token][9216] bf16 = [silu(x) | bases].
// Each (token, feature) evaluated exactly once.
// ---------------------------------------------------------------------------
__global__ __launch_bounds__(256) void act_kernel(
    const float* __restrict__ x, bf16* __restrict__ actF)
{
  const int tok  = blockIdx.x;
  const int feat = blockIdx.y * 256 + threadIdx.x;
  float xv = x[(size_t)tok * HIDDEN + feat];
  bf16* row = actF + (size_t)tok * KACT;
  row[feat] = (bf16)silu(xv);
  float bb[8];
  bspline8(xv, bb);
  v8bf p;
#pragma unroll
  for (int j = 0; j < 8; ++j) p[j] = (bf16)bb[j];
  *reinterpret_cast<v8bf*>(&row[HIDDEN + feat * 8]) = p;
}

// ---------------------------------------------------------------------------
// Kernel 1: qkv = actF @ [base_w | spline_w*scaler]^T.  64x128 tile / block.
// Fragment-ready LDS: AsF[rt][half][l15][e], BsF[col][half][e]
// ---------------------------------------------------------------------------
__global__ __launch_bounds__(256) void kan_qkv_kernel(
    const bf16* __restrict__ actF, const float* __restrict__ bw,
    const float* __restrict__ sw, const float* __restrict__ ss,
    float* __restrict__ qkv)
{
  __shared__ alignas(32) bf16 AsF[64 * 32];    // 4 KB
  __shared__ alignas(32) bf16 BsF[128 * 32];   // 8 KB
  const int mb = blockIdx.x * 64;
  const int nb = blockIdx.y * 128;
  const int tid = threadIdx.x;
  const int lane = tid & 31, wv = tid >> 5;
  const int half = lane >> 4, l15 = lane & 15;
  const int rt = wv & 3;
  const int ct0 = (wv >> 2) * 4;

  const int a_frag = ((rt * 2 + half) * 16 + l15) * 16;
  int b_frag[4];
#pragma unroll
  for (int t = 0; t < 4; ++t)
    b_frag[t] = (((ct0 + t) * 16 + l15) * 2 + half) * 16;

  // staging indices
  const int s_row = tid >> 2, s_sub = tid & 3;        // A: 8 cols each
  const int a_dst = (((s_row >> 4) * 2 + (s_sub >> 1)) * 16 + (s_row & 15)) * 16
                    + (s_sub & 1) * 8;
  const int s_nn = tid & 127, s_k0 = (tid >> 7) * 16; // B: 16 consecutive k
  const int b_dst = (s_nn * 2 + (s_k0 >> 4)) * 16;

  v8f c[4];
#pragma unroll
  for (int t = 0; t < 4; ++t) c[t] = (v8f){};

  const bf16* arow_src = actF + (size_t)(mb + s_row) * KACT + s_sub * 8;

  for (int kc = 0; kc < 288; ++kc) {
    // ---- stage A: pure bf16 copy from precomputed activations ----
    *reinterpret_cast<v8bf*>(&AsF[a_dst]) =
        *reinterpret_cast<const v8bf*>(arow_src + kc * 32);
    // ---- stage B: 16 consecutive K -> one v16bf run ----
    {
      int n = nb + s_nn;
      v8f w0v, w1v;
      if (kc < 32) {
        const float* src = &bw[(size_t)n * HIDDEN + kc * 32 + s_k0];
        w0v = *(const v8f*)src;
        w1v = *(const v8f*)(src + 8);
      } else {
        int kk0 = (kc - 32) * 32 + s_k0;             // multiple of 8
        const float* src = &sw[(size_t)n * 8192 + kk0];
        float sc0 = ss[(size_t)n * HIDDEN + (kk0 >> 3)];
        float sc1 = ss[(size_t)n * HIDDEN + (kk0 >> 3) + 1];
        w0v = *(const v8f*)src;
        w1v = *(const v8f*)(src + 8);
#pragma unroll
        for (int j = 0; j < 8; ++j) { w0v[j] *= sc0; w1v[j] *= sc1; }
        if (kc < 287) __builtin_prefetch(&sw[(size_t)n * 8192 + kk0 + 32], 0, 1);
      }
      *reinterpret_cast<v16bf*>(&BsF[b_dst]) = cat16(cvt8(w0v), cvt8(w1v));
    }
    __syncthreads();

    v16bf a = *reinterpret_cast<const v16bf*>(&AsF[a_frag]);
#pragma unroll
    for (int t = 0; t < 4; ++t) {
      v16bf bv = *reinterpret_cast<const v16bf*>(&BsF[b_frag[t]]);
      c[t] = wmma_bf16(a, bv, c[t]);
    }
    __syncthreads();
  }

  int n0 = nb + ct0 * 16 + l15;
#pragma unroll
  for (int r = 0; r < 8; ++r) {
    int row = mb + rt * 16 + r + 8 * half;
#pragma unroll
    for (int t = 0; t < 4; ++t)
      qkv[(size_t)row * NOUT + n0 + t * 16] = c[t][r];
  }
}

// ---------------------------------------------------------------------------
// Kernel 2: qr/kr[bh][s][m] = sum_d q[s,d]*R[m,d]; q scaled by 1/8.
// ---------------------------------------------------------------------------
__global__ __launch_bounds__(256) void rope_kernel(
    const float* __restrict__ qkv, const float* __restrict__ R,
    bf16* __restrict__ qr, bf16* __restrict__ kr)
{
  const int bh = blockIdx.y;
  const int b = bh >> 4, h = bh & 15;
  const int sblk = blockIdx.x * 64;
  const int tid = threadIdx.x, lane = tid & 31, wv = tid >> 5;
  const int half = lane >> 4, l15 = lane & 15;
  const int isK = wv >> 2;
  const int rt = wv & 3;
  const int off = isK ? 64 : 0;
  bf16* dst = isK ? kr : qr;
  const float scale = isK ? 1.0f : 0.125f;

  v16bf a0, a1;
  {
    int srow = sblk + rt * 16 + l15;
    const float* qp = qkv + ((size_t)b * SEQ + srow) * NOUT + h * 192 + off;
    a0 = cat16(cvt8(*(const v8f*)(qp + 8 * half)),
               cvt8(*(const v8f*)(qp + 16 + 8 * half)));
    a1 = cat16(cvt8(*(const v8f*)(qp + 32 + 8 * half)),
               cvt8(*(const v8f*)(qp + 48 + 8 * half)));
  }
#pragma unroll
  for (int ct = 0; ct < 4; ++ct) {
    int m = ct * 16 + l15;
    const float* rp = R + m * HDIM;
    v16bf b0 = cat16(cvt8(*(const v8f*)(rp + 8 * half)),
                     cvt8(*(const v8f*)(rp + 16 + 8 * half)));
    v16bf b1 = cat16(cvt8(*(const v8f*)(rp + 32 + 8 * half)),
                     cvt8(*(const v8f*)(rp + 48 + 8 * half)));
    v8f cc = {};
    cc = wmma_bf16(a0, b0, cc);
    cc = wmma_bf16(a1, b1, cc);
#pragma unroll
    for (int r = 0; r < 8; ++r) {
      int s = sblk + rt * 16 + r + 8 * half;
      dst[((size_t)bh * SEQ + s) * HDIM + m] = (bf16)(cc[r] * scale);
    }
  }
}

// ---------------------------------------------------------------------------
// Kernel 3: flash attention. Scores transposed so lane p-registers ARE the
// PV A-fragment. V tile staged once per block in V^T[d][key] bf16 layout.
// ---------------------------------------------------------------------------
__global__ __launch_bounds__(256) void attn_kernel(
    const bf16* __restrict__ qr, const bf16* __restrict__ kr,
    const float* __restrict__ qkv, float* __restrict__ ctx)
{
  __shared__ alignas(32) bf16 VsF[64 * 32];  // [d][key]
  const int bh = blockIdx.y, b = bh >> 4, h = bh & 15;
  const int tid = threadIdx.x, lane = tid & 31, wv = tid >> 5;
  const int half = lane >> 4, l15 = lane & 15;
  const int qs = blockIdx.x * 128 + wv * 16;

  v16bf bq0, bq1;
  {
    const bf16* qp = qr + ((size_t)bh * SEQ + qs + l15) * HDIM;
    bq0 = cat16(*(const v8bf*)(qp + 8 * half), *(const v8bf*)(qp + 16 + 8 * half));
    bq1 = cat16(*(const v8bf*)(qp + 32 + 8 * half), *(const v8bf*)(qp + 48 + 8 * half));
  }

  v8f acc[4];
#pragma unroll
  for (int dc = 0; dc < 4; ++dc) acc[dc] = (v8f){};
  float m = -1e30f, l = 0.0f;

  const int v_key = tid >> 3;
  const int v_d0  = (tid & 7) * 8;

  for (int kb = 0; kb < 64; ++kb) {
    int t0 = kb * 32;
    {
      const float* vsrc =
          qkv + ((size_t)b * SEQ + t0 + v_key) * NOUT + h * 192 + 128 + v_d0;
      v8f vv = *(const v8f*)vsrc;
#pragma unroll
      for (int j = 0; j < 8; ++j) VsF[(v_d0 + j) * 32 + v_key] = (bf16)vv[j];
    }
    v8f s0 = {}, s1 = {};
    {
      const bf16* kp0 = kr + ((size_t)bh * SEQ + t0 + l15) * HDIM;
      const bf16* kp1 = kp0 + 16 * HDIM;
      v16bf ak0a = cat16(*(const v8bf*)(kp0 + 8 * half),
                         *(const v8bf*)(kp0 + 16 + 8 * half));
      v16bf ak0b = cat16(*(const v8bf*)(kp0 + 32 + 8 * half),
                         *(const v8bf*)(kp0 + 48 + 8 * half));
      v16bf ak1a = cat16(*(const v8bf*)(kp1 + 8 * half),
                         *(const v8bf*)(kp1 + 16 + 8 * half));
      v16bf ak1b = cat16(*(const v8bf*)(kp1 + 32 + 8 * half),
                         *(const v8bf*)(kp1 + 48 + 8 * half));
      s0 = wmma_bf16(ak0a, bq0, s0);  s0 = wmma_bf16(ak0b, bq1, s0);
      s1 = wmma_bf16(ak1a, bq0, s1);  s1 = wmma_bf16(ak1b, bq1, s1);
    }
    float lmax = s0[0];
#pragma unroll
    for (int r = 1; r < 8; ++r) lmax = fmaxf(lmax, s0[r]);
#pragma unroll
    for (int r = 0; r < 8; ++r) lmax = fmaxf(lmax, s1[r]);
    float bmax = fmaxf(lmax, __shfl_xor(lmax, 16, 32));
    float mn = fmaxf(m, bmax);
    float alpha = __expf(m - mn);
    float p0[8], p1[8], lsum = 0.0f;
#pragma unroll
    for (int r = 0; r < 8; ++r) {
      p0[r] = __expf(s0[r] - mn);
      p1[r] = __expf(s1[r] - mn);
      lsum += p0[r] + p1[r];
    }
    l = l * alpha + lsum + __shfl_xor(lsum, 16, 32);
    m = mn;
#pragma unroll
    for (int r = 0; r < 8; ++r) {
      float ar = __shfl(alpha, r + 8 * half, 32);
      acc[0][r] *= ar; acc[1][r] *= ar; acc[2][r] *= ar; acc[3][r] *= ar;
    }
    v16bf ap;
#pragma unroll
    for (int e = 0; e < 8; ++e)  ap[e] = (bf16)p0[e];
#pragma unroll
    for (int e = 8; e < 16; ++e) ap[e] = (bf16)p1[e - 8];

    __syncthreads();
#pragma unroll
    for (int dc = 0; dc < 4; ++dc) {
      v16bf bv = cat16(*reinterpret_cast<const v8bf*>(&VsF[(dc * 16 + l15) * 32 + 8 * half]),
          *reinterpret_cast<const v8bf*>(&VsF[(dc * 16 + l15) * 32 + 16 + 8 * half]));
      acc[dc] = wmma_bf16(ap, bv, acc[dc]);
    }
    __syncthreads();
  }
#pragma unroll
  for (int r = 0; r < 8; ++r) {
    float lr = __shfl(l, r + 8 * half, 32);
    float inv = 1.0f / lr;
    int s = qs + r + 8 * half;
    size_t tok = (size_t)b * SEQ + s;
#pragma unroll
    for (int dc = 0; dc < 4; ++dc)
      ctx[tok * HIDDEN + h * HDIM + dc * 16 + l15] = acc[dc][r] * inv;
  }
}

// ---------------------------------------------------------------------------
// Kernel 4: out = ctx @ out_w^T + out_b.  64x128 tile / block.
// ---------------------------------------------------------------------------
__global__ __launch_bounds__(256) void proj_kernel(
    const float* __restrict__ ctx, const float* __restrict__ W,
    const float* __restrict__ bias, float* __restrict__ out)
{
  __shared__ alignas(32) bf16 AsF[64 * 32];
  __shared__ alignas(32) bf16 BsF[128 * 32];
  const int mb = blockIdx.x * 64, nb = blockIdx.y * 128;
  const int tid = threadIdx.x, lane = tid & 31, wv = tid >> 5;
  const int half = lane >> 4, l15 = lane & 15;
  const int rt = wv & 3, ct0 = (wv >> 2) * 4;

  const int a_frag = ((rt * 2 + half) * 16 + l15) * 16;
  int b_frag[4];
#pragma unroll
  for (int t = 0; t < 4; ++t)
    b_frag[t] = (((ct0 + t) * 16 + l15) * 2 + half) * 16;
  const int s_row = tid >> 2, s_sub = tid & 3;
  const int a_dst = (((s_row >> 4) * 2 + (s_sub >> 1)) * 16 + (s_row & 15)) * 16
                    + (s_sub & 1) * 8;
  const int s_nn = tid & 127, s_k0 = (tid >> 7) * 16;
  const int b_dst = (s_nn * 2 + (s_k0 >> 4)) * 16;

  v8f c[4];
#pragma unroll
  for (int t = 0; t < 4; ++t) c[t] = (v8f){};

  for (int kc = 0; kc < 32; ++kc) {
    {
      v8f v = *(const v8f*)&ctx[(size_t)(mb + s_row) * HIDDEN + kc * 32 + s_sub * 8];
      *reinterpret_cast<v8bf*>(&AsF[a_dst]) = cvt8(v);
    }
    {
      const float* src = &W[(size_t)(nb + s_nn) * HIDDEN + kc * 32 + s_k0];
      v8f w0v = *(const v8f*)src;
      v8f w1v = *(const v8f*)(src + 8);
      *reinterpret_cast<v16bf*>(&BsF[b_dst]) = cat16(cvt8(w0v), cvt8(w1v));
    }
    __syncthreads();
    v16bf a = *reinterpret_cast<const v16bf*>(&AsF[a_frag]);
#pragma unroll
    for (int t = 0; t < 4; ++t) {
      v16bf bv = *reinterpret_cast<const v16bf*>(&BsF[b_frag[t]]);
      c[t] = wmma_bf16(a, bv, c[t]);
    }
    __syncthreads();
  }
  int n0 = nb + ct0 * 16 + l15;
#pragma unroll
  for (int t = 0; t < 4; ++t) {
    float bi = bias[n0 + t * 16];
#pragma unroll
    for (int r = 0; r < 8; ++r) {
      int row = mb + rt * 16 + r + 8 * half;
      out[(size_t)row * HIDDEN + n0 + t * 16] = c[t][r] + bi;
    }
  }
}

// ---------------------------------------------------------------------------
extern "C" void kernel_launch(void* const* d_in, const int* in_sizes, int n_in,
                              void* d_out, int out_size, void* d_ws, size_t ws_size,
                              hipStream_t stream) {
  (void)in_sizes; (void)n_in; (void)out_size; (void)ws_size;
  const float* x  = (const float*)d_in[0];   // (2,2048,1024)
  const float* bw = (const float*)d_in[1];   // (3072,1024)
  const float* sw = (const float*)d_in[2];   // (3072,1024,8)
  const float* ss = (const float*)d_in[3];   // (3072,1024)
  // d_in[4] = grid: uniform, knots hardcoded (t_m = 0.4m - 2.2)
  const float* R  = (const float*)d_in[5];   // (64,64)
  const float* W  = (const float*)d_in[6];   // (1024,1024)
  const float* bb = (const float*)d_in[7];   // (1024,)
  float* out = (float*)d_out;

  char* ws = (char*)d_ws;
  const size_t MB = 1024 * 1024;
  bf16*  actF = (bf16*)ws;                       // 4096*9216*2 = 72 MiB
  float* qkv  = (float*)(ws + 72 * MB);          // 48 MiB
  bf16*  qr   = (bf16*)(ws + 120 * MB);          // 8 MiB
  bf16*  kr   = (bf16*)(ws + 128 * MB);          // 8 MiB
  float* ctx  = (float*)(ws + 136 * MB);         // 16 MiB

  act_kernel<<<dim3(NTOK, HIDDEN / 256), 256, 0, stream>>>(x, actF);
  kan_qkv_kernel<<<dim3(NTOK / 64, NOUT / 128), 256, 0, stream>>>(actF, bw, sw, ss, qkv);
  rope_kernel<<<dim3(SEQ / 64, 32), 256, 0, stream>>>(qkv, R, qr, kr);
  attn_kernel<<<dim3(SEQ / 128, 32), 256, 0, stream>>>(qr, kr, qkv, ctx);
  proj_kernel<<<dim3(NTOK / 64, HIDDEN / 128), 256, 0, stream>>>(ctx, W, bb, out);
}
